// EnhancedBrawlerPredictionModel_13134009991543
// MI455X (gfx1250) — matrix operand order repaired and checked by reading, not compiled, not run on hardware
//
#include <hip/hip_runtime.h>

typedef __attribute__((ext_vector_type(16))) _Float16 v16h;
typedef __attribute__((ext_vector_type(8)))  float    v8f;

#define E       32
#define H1      128
#define H2      64
#define NOUT    512
#define INDIM   224      // 7 * 32
#define WSTRIDE 33       // LDS row stride (pad to avoid bank conflicts)
#define MHA_SZ  4352     // 96*33 + 96 + 32*33 + 32

// ---------------------------------------------------------------------------
// Front-end: one wave (32 lanes) per sample, lane == embedding dimension.
// ---------------------------------------------------------------------------
__device__ __forceinline__ void mha32(const float xq[3], const float xkv[3],
                                      const float* __restrict__ w, int lane,
                                      float out[3])
{
    const float* iw = w;            // 96 rows, stride 33
    const float* ib = w + 3168;     // 96
    const float* ow = w + 3264;     // 32 rows, stride 33
    const float* ob = w + 4320;     // 32

    const float* rq = iw + lane * WSTRIDE;
    const float* rk = iw + (E + lane) * WSTRIDE;
    const float* rv = iw + (2 * E + lane) * WSTRIDE;

    float q[3], k[3], v[3];
    for (int t = 0; t < 3; ++t) {
        float aq = 0.f, ak = 0.f, av = 0.f;
        for (int i = 0; i < E; ++i) {
            float xqi = __shfl(xq[t], i, 32);
            float xki = __shfl(xkv[t], i, 32);
            aq = fmaf(xqi, rq[i], aq);
            ak = fmaf(xki, rk[i], ak);
            av = fmaf(xki, rv[i], av);
        }
        q[t] = aq + ib[lane];
        k[t] = ak + ib[E + lane];
        v[t] = av + ib[2 * E + lane];
    }

    // scores per head: reduce over the 8 lanes of each head (dh = 8)
    float sc[3][3];
    for (int t1 = 0; t1 < 3; ++t1)
        for (int t2 = 0; t2 < 3; ++t2) {
            float p = q[t1] * k[t2];
            p += __shfl_xor(p, 1, 32);
            p += __shfl_xor(p, 2, 32);
            p += __shfl_xor(p, 4, 32);
            sc[t1][t2] = p * 0.35355339059327373f;   // 1/sqrt(8)
        }

    float ao[3];
    for (int t1 = 0; t1 < 3; ++t1) {
        float mx = fmaxf(fmaxf(sc[t1][0], sc[t1][1]), sc[t1][2]);
        float e0 = __expf(sc[t1][0] - mx);
        float e1 = __expf(sc[t1][1] - mx);
        float e2 = __expf(sc[t1][2] - mx);
        float inv = 1.0f / (e0 + e1 + e2);
        ao[t1] = (e0 * v[0] + e1 * v[1] + e2 * v[2]) * inv;
    }

    const float* ro = ow + lane * WSTRIDE;
    for (int t = 0; t < 3; ++t) {
        float a = 0.f;
        for (int i = 0; i < E; ++i)
            a = fmaf(__shfl(ao[t], i, 32), ro[i], a);
        out[t] = a + ob[lane];
    }
}

__global__ void __launch_bounds__(256)
k_frontend(const int* __restrict__ friends, const int* __restrict__ enemies,
           const int* __restrict__ map_idx,
           const float* __restrict__ bemb, const float* __restrict__ memb,
           const float* __restrict__ pos_w, const float* __restrict__ pos_b,
           const float* __restrict__ ea_iw, const float* __restrict__ ea_ib,
           const float* __restrict__ ea_ow, const float* __restrict__ ea_ob,
           const float* __restrict__ fa_iw, const float* __restrict__ fa_ib,
           const float* __restrict__ fa_ow, const float* __restrict__ fa_ob,
           const float* __restrict__ ca_iw, const float* __restrict__ ca_ib,
           const float* __restrict__ ca_ow, const float* __restrict__ ca_ob,
           _Float16* __restrict__ X)
{
    __shared__ float sw[3 * MHA_SZ];

    const float* iws[3] = { ea_iw, fa_iw, ca_iw };
    const float* ibs[3] = { ea_ib, fa_ib, ca_ib };
    const float* ows[3] = { ea_ow, fa_ow, ca_ow };
    const float* obs[3] = { ea_ob, fa_ob, ca_ob };

    for (int m = 0; m < 3; ++m) {
        float* base = sw + m * MHA_SZ;
        for (int idx = threadIdx.x; idx < 96 * 32; idx += blockDim.x)
            base[(idx >> 5) * WSTRIDE + (idx & 31)] = iws[m][idx];
        for (int idx = threadIdx.x; idx < 96; idx += blockDim.x)
            base[3168 + idx] = ibs[m][idx];
        for (int idx = threadIdx.x; idx < 32 * 32; idx += blockDim.x)
            base[3264 + (idx >> 5) * WSTRIDE + (idx & 31)] = ows[m][idx];
        for (int idx = threadIdx.x; idx < 32; idx += blockDim.x)
            base[4320 + idx] = obs[m][idx];
    }
    __syncthreads();

    int wave = threadIdx.x >> 5;
    int lane = threadIdx.x & 31;
    int s = blockIdx.x * 8 + wave;

    float pw = pos_w[lane], pb = pos_b[lane];

    float ef[3], ae[3];
    for (int j = 0; j < 3; ++j) {
        int fb = friends[s * 3 + j];
        ef[j] = bemb[fb * E + lane];
        int eb = enemies[s * 3 + j];
        ae[j] = bemb[eb * E + lane] + (float)j * pw + pb;
    }

    float ea_o[3], fa_o[3], ca_o[3];
    mha32(ae, ae, sw + 0 * MHA_SZ, lane, ea_o);
    mha32(ef, ef, sw + 1 * MHA_SZ, lane, fa_o);
    mha32(fa_o, ea_o, sw + 2 * MHA_SZ, lane, ca_o);   // q = fa, k/v = ea

    int mi = map_idx[s];
    float em = memb[mi * E + lane];

    _Float16* xr = X + (size_t)s * INDIM;
    for (int t = 0; t < 3; ++t) xr[t * E + lane]      = (_Float16)ca_o[t];
    for (int t = 0; t < 3; ++t) xr[96 + t * E + lane] = (_Float16)ea_o[t];
    xr[192 + lane] = (_Float16)em;
}

// ---------------------------------------------------------------------------
// One-shot f32 -> f16 weight conversion (weights are batch-invariant).
// ---------------------------------------------------------------------------
__global__ void k_prep(const float* __restrict__ w1, const float* __restrict__ w2,
                       const float* __restrict__ w3,
                       _Float16* __restrict__ h1, _Float16* __restrict__ h2,
                       _Float16* __restrict__ h3)
{
    int i = blockIdx.x * blockDim.x + threadIdx.x;
    if (i < H1 * INDIM)  h1[i] = (_Float16)w1[i];
    if (i < H2 * H1)     h2[i] = (_Float16)w2[i];
    if (i < NOUT * H2)   h3[i] = (_Float16)w3[i];
}

// ---------------------------------------------------------------------------
// WMMA GEMM kernels.  A fragment (16x32 f16): lanes 0-15 -> M = lane, K in
// {hi*8 .. hi*8+7} and {16+hi*8 .. 23+hi*8} (hi = lane>>4).
// B fragment (32x16 f16): N = lane&15, K same split as A.
// C/D: VGPR r -> row r + 8*hi, col lane&15.
// Each block covers M=64 rows; B fragments hoisted into registers and reused
// across 4 M-subtiles.
// ---------------------------------------------------------------------------
__device__ __forceinline__ v8f wmma_step(v16h a, v16h b, v8f c)
{
    return __builtin_amdgcn_wmma_f32_16x16x32_f16(false, a, false, b,
                                                  (short)0, c, false, false);
}

__device__ __forceinline__ v16h load_afrag(const _Float16* p)
{
    v16h a;
    for (int j = 0; j < 8; ++j) { a[j] = p[j]; a[8 + j] = p[16 + j]; }
    return a;
}

__device__ __forceinline__ v16h load_bfrag(const _Float16* p)
{
    v16h b;
    for (int j = 0; j < 8; ++j) { b[j] = p[j]; b[8 + j] = p[16 + j]; }
    return b;
}

__global__ void __launch_bounds__(256)
k_fc1(const _Float16* __restrict__ X, const _Float16* __restrict__ Wh,
      const float* __restrict__ bias, float* __restrict__ Y,
      float* __restrict__ ssum, float* __restrict__ ssq)
{
    __shared__ __align__(16) _Float16 sA[64 * INDIM];   // 28 KB
    int m0 = blockIdx.x * 64;

    {   // linear 128-bit staging copy (rows are contiguous in X)
        const uint4* src = (const uint4*)(X + (size_t)m0 * INDIM);
        uint4* dst = (uint4*)sA;
        for (int idx = threadIdx.x; idx < 64 * INDIM / 8; idx += 256)
            dst[idx] = src[idx];
    }
    __syncthreads();

    int wave = threadIdx.x >> 5, lane = threadIdx.x & 31;
    int mrow = lane & 15, hi = lane >> 4;
    int n = wave * 16 + mrow;

    v16h bfr[7];
    for (int ks = 0; ks < 7; ++ks)
        bfr[ks] = load_bfrag(Wh + (size_t)n * INDIM + ks * 32 + hi * 8);

    float bn = bias[n];
    float lsum = 0.f, lsq = 0.f;

    for (int mt = 0; mt < 4; ++mt) {
        const _Float16* arow = sA + (mt * 16 + mrow) * INDIM + hi * 8;
        v8f acc = {};
        for (int ks = 0; ks < 7; ++ks)
            acc = wmma_step(load_afrag(arow + ks * 32), bfr[ks], acc);
        for (int r = 0; r < 8; ++r) {
            float y = acc[r] + bn;
            Y[(size_t)(m0 + mt * 16 + r + 8 * hi) * H1 + n] = y;
            lsum += y; lsq += y * y;
        }
    }
    lsum += __shfl_xor(lsum, 16, 32);
    lsq  += __shfl_xor(lsq, 16, 32);
    if (lane < 16) {
        atomicAdd(&ssum[n], lsum);
        atomicAdd(&ssq[n],  lsq);
    }
}

__global__ void __launch_bounds__(128)
k_fc2(const float* __restrict__ Y, const _Float16* __restrict__ Wh,
      const float* __restrict__ bias,
      const float* __restrict__ sc1, const float* __restrict__ sh1,
      float* __restrict__ Z, float* __restrict__ ssum, float* __restrict__ ssq)
{
    __shared__ __align__(16) _Float16 sA[64 * H1];      // 16 KB
    int m0 = blockIdx.x * 64;

    for (int idx = threadIdx.x; idx < 64 * H1; idx += 128) {
        int k = idx & 127;
        float h = Y[(size_t)m0 * H1 + idx] * sc1[k] + sh1[k];
        sA[idx] = (_Float16)(h > 0.f ? h : 0.f);
    }
    __syncthreads();

    int wave = threadIdx.x >> 5, lane = threadIdx.x & 31;
    int mrow = lane & 15, hi = lane >> 4;
    int n = wave * 16 + mrow;

    v16h bfr[4];
    for (int ks = 0; ks < 4; ++ks)
        bfr[ks] = load_bfrag(Wh + (size_t)n * H1 + ks * 32 + hi * 8);

    float bn = bias[n];
    float lsum = 0.f, lsq = 0.f;

    for (int mt = 0; mt < 4; ++mt) {
        const _Float16* arow = sA + (mt * 16 + mrow) * H1 + hi * 8;
        v8f acc = {};
        for (int ks = 0; ks < 4; ++ks)
            acc = wmma_step(load_afrag(arow + ks * 32), bfr[ks], acc);
        for (int r = 0; r < 8; ++r) {
            float z = acc[r] + bn;
            Z[(size_t)(m0 + mt * 16 + r + 8 * hi) * H2 + n] = z;
            lsum += z; lsq += z * z;
        }
    }
    lsum += __shfl_xor(lsum, 16, 32);
    lsq  += __shfl_xor(lsq, 16, 32);
    if (lane < 16) {
        atomicAdd(&ssum[n], lsum);
        atomicAdd(&ssq[n],  lsq);
    }
}

__global__ void __launch_bounds__(512)
k_fc3(const float* __restrict__ Z, const _Float16* __restrict__ Wh,
      const float* __restrict__ bias,
      const float* __restrict__ sc2, const float* __restrict__ sh2,
      const int* __restrict__ enemies, const float* __restrict__ cm,
      float* __restrict__ out)
{
    __shared__ __align__(16) _Float16 sA[64 * H2];      // 8 KB
    __shared__ int   sE[64][3];
    __shared__ float sScale[64];

    int m0 = blockIdx.x * 64;
    for (int idx = threadIdx.x; idx < 64 * H2; idx += 512) {
        int k = idx & 63;
        float h = Z[(size_t)m0 * H2 + idx] * sc2[k] + sh2[k];
        sA[idx] = (_Float16)(h > 0.f ? h : 0.f);
    }
    if (threadIdx.x < 64) {
        int e0 = enemies[(m0 + threadIdx.x) * 3 + 0];
        int e1 = enemies[(m0 + threadIdx.x) * 3 + 1];
        int e2 = enemies[(m0 + threadIdx.x) * 3 + 2];
        sE[threadIdx.x][0] = e0; sE[threadIdx.x][1] = e1; sE[threadIdx.x][2] = e2;
        int valid = (e0 != 0) + (e1 != 0) + (e2 != 0);
        sScale[threadIdx.x] = valid > 0 ? 1.0f / (float)valid : 0.0f;
    }
    __syncthreads();

    int wave = threadIdx.x >> 5, lane = threadIdx.x & 31;
    int mrow = lane & 15, hi = lane >> 4;

    // B fragments: 2 N-tiles x 2 K-steps, hoisted
    v16h bfr[2][2];
    float bn[2];
    for (int nt = 0; nt < 2; ++nt) {
        int n = wave * 32 + nt * 16 + mrow;
        const _Float16* pwv = Wh + (size_t)n * H2 + hi * 8;
        bfr[nt][0] = load_bfrag(pwv);
        bfr[nt][1] = load_bfrag(pwv + 32);
        bn[nt] = bias[n];
    }

    for (int mt = 0; mt < 4; ++mt) {
        const _Float16* arow = sA + (mt * 16 + mrow) * H2 + hi * 8;
        v16h a0 = load_afrag(arow);
        v16h a1 = load_afrag(arow + 32);
        for (int nt = 0; nt < 2; ++nt) {
            int n = wave * 32 + nt * 16 + mrow;
            v8f acc = {};
            acc = wmma_step(a0, bfr[nt][0], acc);
            acc = wmma_step(a1, bfr[nt][1], acc);
            for (int r = 0; r < 8; ++r) {
                int mloc = mt * 16 + r + 8 * hi;
                float infl = 0.f;
                int e0 = sE[mloc][0], e1 = sE[mloc][1], e2 = sE[mloc][2];
                if (e0 != 0) infl += cm[(size_t)e0 * NOUT + n];
                if (e1 != 0) infl += cm[(size_t)e1 * NOUT + n];
                if (e2 != 0) infl += cm[(size_t)e2 * NOUT + n];
                out[(size_t)(m0 + mloc) * NOUT + n] =
                    acc[r] + bn[nt] + sScale[mloc] * infl;
            }
        }
    }
}

// ---------------------------------------------------------------------------
__global__ void k_init(float* st, int n)
{
    int i = blockIdx.x * blockDim.x + threadIdx.x;
    if (i < n) st[i] = 0.0f;
}

__global__ void k_bnfin(const float* __restrict__ ssum, const float* __restrict__ ssq,
                        const float* __restrict__ g, const float* __restrict__ b,
                        float* __restrict__ scale, float* __restrict__ shift,
                        int C, float invB)
{
    int c = blockIdx.x * blockDim.x + threadIdx.x;
    if (c < C) {
        float m   = ssum[c] * invB;
        float var = ssq[c] * invB - m * m;
        float s   = rsqrtf(var + 1e-5f) * g[c];
        scale[c] = s;
        shift[c] = b[c] - m * s;
    }
}

// ---------------------------------------------------------------------------
extern "C" void kernel_launch(void* const* d_in, const int* in_sizes, int n_in,
                              void* d_out, int out_size, void* d_ws, size_t ws_size,
                              hipStream_t stream)
{
    (void)n_in; (void)out_size; (void)ws_size;
    const int* friends = (const int*)d_in[0];
    const int* enemies = (const int*)d_in[1];
    const int* map_idx = (const int*)d_in[2];
    const float* bemb  = (const float*)d_in[3];
    const float* memb  = (const float*)d_in[4];
    const float* cm    = (const float*)d_in[5];
    const float* pos_w = (const float*)d_in[6];
    const float* pos_b = (const float*)d_in[7];
    const float* fc1_w = (const float*)d_in[8];
    const float* fc1_b = (const float*)d_in[9];
    const float* bn1_g = (const float*)d_in[10];
    const float* bn1_b = (const float*)d_in[11];
    const float* fc2_w = (const float*)d_in[12];
    const float* fc2_b = (const float*)d_in[13];
    const float* bn2_g = (const float*)d_in[14];
    const float* bn2_b = (const float*)d_in[15];
    const float* fc3_w = (const float*)d_in[16];
    const float* fc3_b = (const float*)d_in[17];
    const float* ea_iw = (const float*)d_in[18];
    const float* ea_ib = (const float*)d_in[19];
    const float* ea_ow = (const float*)d_in[20];
    const float* ea_ob = (const float*)d_in[21];
    const float* fa_iw = (const float*)d_in[22];
    const float* fa_ib = (const float*)d_in[23];
    const float* fa_ow = (const float*)d_in[24];
    const float* fa_ob = (const float*)d_in[25];
    const float* ca_iw = (const float*)d_in[26];
    const float* ca_ib = (const float*)d_in[27];
    const float* ca_ow = (const float*)d_in[28];
    const float* ca_ob = (const float*)d_in[29];

    const int B = in_sizes[0] / 3;   // 65536

    // workspace layout (bytes)
    char* ws = (char*)d_ws;
    size_t offX  = 0;
    size_t offY1 = offX  + (size_t)B * INDIM * 2;        // X : B*224 f16
    size_t offZ  = offY1 + (size_t)B * H1 * 4;           // Y1: B*128 f32
    size_t offSt = offZ  + (size_t)B * H2 * 4;           // Z : B*64  f32
    size_t offW  = offSt + 4096;                         // stats block
    _Float16* X  = (_Float16*)(ws + offX);
    float*    Y1 = (float*)(ws + offY1);
    float*    Z  = (float*)(ws + offZ);
    float*    st = (float*)(ws + offSt);
    _Float16* Wh1 = (_Float16*)(ws + offW);                          // 128*224
    _Float16* Wh2 = (_Float16*)(ws + offW + H1 * INDIM * 2);         // 64*128
    _Float16* Wh3 = (_Float16*)(ws + offW + H1 * INDIM * 2 + H2 * H1 * 2); // 512*64

    float* ssum1 = st;          // 128
    float* ssq1  = st + 128;    // 128
    float* ssum2 = st + 256;    // 64
    float* ssq2  = st + 320;    // 64
    float* sc1   = st + 384;    // 128
    float* sh1   = st + 512;    // 128
    float* sc2   = st + 640;    // 64
    float* sh2   = st + 704;    // 64

    float invB = 1.0f / (float)B;

    k_init<<<1, 384, 0, stream>>>(st, 384);
    k_prep<<<(NOUT * H2 + 255) / 256, 256, 0, stream>>>(fc1_w, fc2_w, fc3_w,
                                                        Wh1, Wh2, Wh3);

    k_frontend<<<B / 8, 256, 0, stream>>>(
        friends, enemies, map_idx, bemb, memb, pos_w, pos_b,
        ea_iw, ea_ib, ea_ow, ea_ob,
        fa_iw, fa_ib, fa_ow, fa_ob,
        ca_iw, ca_ib, ca_ow, ca_ob, X);

    k_fc1<<<B / 64, 256, 0, stream>>>(X, Wh1, fc1_b, Y1, ssum1, ssq1);
    k_bnfin<<<1, 128, 0, stream>>>(ssum1, ssq1, bn1_g, bn1_b, sc1, sh1, H1, invB);
    k_fc2<<<B / 64, 128, 0, stream>>>(Y1, Wh2, fc2_b, sc1, sh1, Z, ssum2, ssq2);
    k_bnfin<<<1, 64, 0, stream>>>(ssum2, ssq2, bn2_g, bn2_b, sc2, sh2, H2, invB);
    k_fc3<<<B / 64, 512, 0, stream>>>(Z, Wh3, fc3_b, sc2, sh2, enemies, cm,
                                      (float*)d_out);
}
